// HetSTGCNBlock_55448027791678
// MI455X (gfx1250) — matrix operand order repaired and checked by reading, not compiled, not run on hardware
//
#include <hip/hip_runtime.h>
#include <cstdint>
#include <cstddef>

// ---------------------------------------------------------------------------
// CDNA5 (gfx1250) HetSTGCN implementation.
// All heavy ops lower to v_wmma_f32_16x16x32_bf16 (wave32, 16x16 tiles).
// Activations/weights stored as bf16; accumulation in f32.
// Node-dim GEMMs stage the B column panel into LDS with the gfx1250 async
// global->LDS DMA path (ASYNCcnt + s_wait_asynccnt), then keep the panel
// register-resident across all M-tiles of the workgroup.
// ---------------------------------------------------------------------------

typedef __attribute__((ext_vector_type(16))) __bf16 v16bf;
typedef __attribute__((ext_vector_type(8)))  float  v8f;
typedef __attribute__((ext_vector_type(4)))  unsigned int u32x4;
typedef __attribute__((ext_vector_type(4)))  int i32x4;
typedef unsigned short u16;

#if defined(__has_builtin)
#if __has_builtin(__builtin_amdgcn_global_load_async_to_lds_b128) && \
    __has_builtin(__builtin_amdgcn_s_wait_asynccnt)
#define USE_ASYNC_LDS 1
#endif
#endif

union ABTile { v16bf v; u32x4 q[2]; u16 h[16]; };

__device__ __forceinline__ u16 f2bf(float f) {
  union { float f; unsigned u; } x; x.f = f;
  unsigned r = x.u + 0x7FFFu + ((x.u >> 16) & 1u);   // round-to-nearest-even
  return (u16)(r >> 16);
}
__device__ __forceinline__ float bf2f(u16 h) {
  union { unsigned u; float f; } x; x.u = ((unsigned)h) << 16;
  return x.f;
}
__device__ __forceinline__ float reluf(float x) { return fmaxf(x, 0.f); }
__device__ __forceinline__ float sigm(float x) { return 1.f / (1.f + __expf(-x)); }

// A-tile (16x32 bf16, row-major source):
// lanes 0-15 hold K = k0+0..15;
// lanes 16-31 hold K = k0+16..31.
__device__ __forceinline__ v16bf load_a(const u16* rowp, int k0, int half) {
  ABTile a;
  const u16* p = rowp + k0 + half * 16;
  a.q[0] = *(const u32x4*)(p);
  a.q[1] = *(const u32x4*)(p + 8);
  return a.v;
}
// B-tile (32x16 bf16) from a pre-transposed weight matrix BT (N x K):
// lane L: column n = L%16; K = k0 + (L<16 ? 0 : 16) + 0..15, contiguous in BT row.
__device__ __forceinline__ v16bf load_bT(const u16* rowp, int k0, int half) {
  ABTile b;
  const u16* p = rowp + k0 + half * 16;
  b.q[0] = *(const u32x4*)(p);
  b.q[1] = *(const u32x4*)(p + 8);
  return b.v;
}
__device__ __forceinline__ v8f wmma_bf16(v16bf a, v16bf b, v8f c) {
  return __builtin_amdgcn_wmma_f32_16x16x32_bf16(false, a, false, b, (short)0, c,
                                                 false, false);
}

// ------------------------------ small prep kernels --------------------------

// dst (Mp x Kp) = pad(src (M x K)) converted to bf16
__global__ void k_cvt_pad(const float* __restrict__ src, u16* __restrict__ dst,
                          int M, int K, int Mp, int Kp) {
  int idx = blockIdx.x * 256 + threadIdx.x;
  if (idx >= Mp * Kp) return;
  int i = idx / Kp, j = idx % Kp;
  dst[idx] = (i < M && j < K) ? f2bf(src[i * K + j]) : (u16)0;
}

// dst (Ncp x Kp) = pad(transpose(src (K x Nc))) -> BT layout for the B operand
__global__ void k_cvt_pad_tr(const float* __restrict__ src, u16* __restrict__ dst,
                             int K, int Nc, int Ncp, int Kp) {
  int idx = blockIdx.x * 256 + threadIdx.x;
  if (idx >= Ncp * Kp) return;
  int n = idx / Kp, k = idx % Kp;
  dst[idx] = (k < K && n < Nc) ? f2bf(src[k * Nc + n]) : (u16)0;
}

// WcatT (192 x 192): fused temporal-conv-2 weights, transposed.
// k = tap*64 + cin ; n<64 -> conv1 out ch, n in 64..191 -> causal out ch.
__global__ void k_build_wcatT(const float* __restrict__ w1,
                              const float* __restrict__ wc,
                              u16* __restrict__ dst) {
  int idx = blockIdx.x * 256 + threadIdx.x;
  if (idx >= 192 * 192) return;
  int n = idx / 192, k = idx % 192;
  int tap = k / 64, ci = k % 64;
  float v = (n < 64) ? w1[n * 192 + ci * 3 + tap]
                     : wc[(n - 64) * 192 + ci * 3 + tap];
  dst[idx] = f2bf(v);
}

// ------------------------- temporal conv 1 (tiny K) -------------------------
// X (16,400,32,2) -> t1 bf16 (16,400,30,64); also scatter into padded u0b/u1b.
__global__ void k_tconv1(const float* __restrict__ X,
                         const float* __restrict__ w1, const float* __restrict__ b1,
                         const float* __restrict__ wc, const float* __restrict__ bc,
                         u16* __restrict__ t1, u16* __restrict__ u0b,
                         u16* __restrict__ u1b) {
  int idx = blockIdx.x * 256 + threadIdx.x;
  if (idx >= 16 * 400 * 30 * 64) return;
  int c = idx & 63;
  int t = (idx >> 6) % 30;
  int n = (idx / (64 * 30)) % 400;
  int b = idx / (64 * 30 * 400);
  const float* xp = X + ((size_t)(b * 400 + n) * 32 + t) * 2;
  float temp = b1[c], p = bc[c], q = bc[64 + c];
#pragma unroll
  for (int tap = 0; tap < 3; ++tap)
#pragma unroll
    for (int i = 0; i < 2; ++i) {
      float xv = xp[tap * 2 + i];
      temp += xv * w1[c * 6 + i * 3 + tap];
      p += xv * wc[c * 6 + i * 3 + tap];
      q += xv * wc[(64 + c) * 6 + i * 3 + tap];
    }
  u16 hv = f2bf((p + temp) * sigm(q));
  t1[idx] = hv;
  if (n < 81)
    u0b[((size_t)(b * 96 + n) * 30 + t) * 64 + c] = hv;
  else
    u1b[((size_t)(b * 320 + (n - 81)) * 30 + t) * 64 + c] = hv;
}

// ----------------- generic GEMM  C(MpxNp) = A(MpxKp) x BT^T ----------------
// (used once to precompute P0 = A01@A10, P1 = A10@A01)
__global__ __launch_bounds__(32) void k_gemm_TB(const u16* __restrict__ A,
                                                const u16* __restrict__ BT,
                                                u16* __restrict__ C,
                                                int Mp, int Kp, int Np) {
  int lane = threadIdx.x & 31, r = lane & 15, half = lane >> 4;
  int m0 = blockIdx.y * 16, c0 = blockIdx.x * 16;
  const u16* arow = A + (size_t)(m0 + r) * Kp;
  const u16* brow = BT + (size_t)(c0 + r) * Kp;
  v8f acc = {};
  for (int k0 = 0; k0 < Kp; k0 += 32)
    acc = wmma_bf16(load_a(arow, k0, half), load_bT(brow, k0, half), acc);
#pragma unroll
  for (int v = 0; v < 8; ++v)
    C[(size_t)(m0 + v + half * 8) * Np + c0 + r] = f2bf(acc[v]);
}

// --------------------- node-dimension GEMM (the bulk) -----------------------
// For each batch b:  S1[b] = G1 x X[b],  S2[b] = G2 x X[b]  (G2 optional),
// X[b] is (KP nodes x 1920) bf16, G* are (Mp x KP) bf16 zero-padded graphs.
// One workgroup owns one 16-column panel: the (KP x 16) B panel is DMA'd
// into LDS once (async global->LDS, ASYNCcnt), read once per wave into
// registers, and reused for all M-tiles (waves split the M dimension).
template <int KP>
__global__ __launch_bounds__(128) void k_node_gemm_lds(
    const u16* __restrict__ G1, const u16* __restrict__ G2,
    const u16* __restrict__ Xa, u16* __restrict__ S1, u16* __restrict__ S2,
    int Mp) {
  constexpr int KS = KP / 32;
  __shared__ u16 bpanel[KP * 16];          // 10.25 KB max (KP=320)

  int wave = threadIdx.x >> 5;
  int lane = threadIdx.x & 31, r = lane & 15, half = lane >> 4;
  int c0 = blockIdx.x * 16;                // 120 column tiles of 1920
  int b = blockIdx.z;

  // ---- stage B panel: KP rows x 32B, coalesced 16B transfers ----
  const u16* gsrc = Xa + (size_t)b * KP * 1920 + c0;
  for (int i = threadIdx.x; i < KP * 2; i += 128) {
    int k = i >> 1, part = (i & 1) * 8;
    const u16* g = gsrc + (size_t)k * 1920 + part;
    u16* l = &bpanel[k * 16 + part];
#ifdef USE_ASYNC_LDS
    __builtin_amdgcn_global_load_async_to_lds_b128((i32x4*)g, (i32x4*)l, 0, 0);
#else
    *(u32x4*)l = *(const u32x4*)g;
#endif
  }
#ifdef USE_ASYNC_LDS
  __builtin_amdgcn_s_wait_asynccnt(0);
#endif
  __syncthreads();

  // ---- pull the panel into register-resident WMMA B fragments ----
  v16bf breg[KS];
#pragma unroll
  for (int ks = 0; ks < KS; ++ks) {
    ABTile bt;
    int kb = ks * 32 + half * 16;
#pragma unroll
    for (int v = 0; v < 8; ++v) {          // conflict-free ds_load_u16
      bt.h[2 * v]     = bpanel[(kb + 2 * v) * 16 + r];
      bt.h[2 * v + 1] = bpanel[(kb + 2 * v + 1) * 16 + r];
    }
    breg[ks] = bt.v;
  }

  // ---- waves split the M-tiles; A-tiles stream from L2-resident graphs ----
  int Mtiles = Mp >> 4;
  u16* s1 = S1 + (size_t)b * Mp * 1920;
  u16* s2 = S2 ? S2 + (size_t)b * Mp * 1920 : (u16*)0;
  for (int mt = wave; mt < Mtiles; mt += 4) {
    int m0 = mt * 16;
    const u16* a1row = G1 + (size_t)(m0 + r) * KP;
    __builtin_prefetch(a1row + (size_t)64 * KP, 0, 1);  // next M-tile of wave
    v8f acc1 = {};
#pragma unroll
    for (int ks = 0; ks < KS; ++ks)
      acc1 = wmma_bf16(load_a(a1row, ks * 32, half), breg[ks], acc1);
#pragma unroll
    for (int v = 0; v < 8; ++v)
      s1[(size_t)(m0 + v + half * 8) * 1920 + c0 + r] = f2bf(acc1[v]);
    if (G2) {
      const u16* a2row = G2 + (size_t)(m0 + r) * KP;
      v8f acc2 = {};
#pragma unroll
      for (int ks = 0; ks < KS; ++ks)
        acc2 = wmma_bf16(load_a(a2row, ks * 32, half), breg[ks], acc2);
#pragma unroll
      for (int v = 0; v < 8; ++v)
        s2[(size_t)(m0 + v + half * 8) * 1920 + c0 + r] = f2bf(acc2[v]);
    }
  }
}

// ------------- fused channel matmuls + het-conv epilogue --------------------
// out = relu(0.9*relu(0.5*(relu(S1@th1) + sigmoid(S2@th2 + S3@th3))) + 0.1*X0)
__global__ __launch_bounds__(128) void k_het_channel(
    const u16* __restrict__ S1, const u16* __restrict__ S2,
    const u16* __restrict__ S3, const u16* __restrict__ th1T,
    const u16* __restrict__ th2T, const u16* __restrict__ th3T,
    const u16* __restrict__ Xres, int nodeBase, u16* __restrict__ Uout,
    int Np, int Nreal) {
  int wave = threadIdx.x >> 5;
  int lane = threadIdx.x & 31, r = lane & 15, half = lane >> 4;
  int c0 = wave * 16;
  int m0 = blockIdx.x * 16;
  const u16* a1 = S1 + (size_t)(m0 + r) * 64;
  const u16* a2 = S2 + (size_t)(m0 + r) * 64;
  const u16* a3 = S3 + (size_t)(m0 + r) * 64;
  const u16* b1r = th1T + (c0 + r) * 64;
  const u16* b2r = th2T + (c0 + r) * 64;
  const u16* b3r = th3T + (c0 + r) * 64;
  v8f A1 = {}, A2 = {}, A3 = {};
#pragma unroll
  for (int k0 = 0; k0 < 64; k0 += 32) {
    A1 = wmma_bf16(load_a(a1, k0, half), load_bT(b1r, k0, half), A1);
    A2 = wmma_bf16(load_a(a2, k0, half), load_bT(b2r, k0, half), A2);
    A3 = wmma_bf16(load_a(a3, k0, half), load_bT(b3r, k0, half), A3);
  }
  int col = c0 + r;
#pragma unroll
  for (int v = 0; v < 8; ++v) {
    int m = m0 + v + half * 8;
    int t = m % 30;
    int node = (m / 30) % Np;
    int b = m / (30 * Np);
    float u2 = reluf(A1[v]);
    float u5 = sigm(A2[v] + A3[v]);
    float u6 = reluf(0.5f * (u2 + u5));
    if (node < Nreal) {     // pad rows stay zero (memset each launch)
      float x0 = bf2f(Xres[((size_t)(b * 400 + nodeBase + node) * 30 + t) * 64 + col]);
      Uout[(size_t)m * 64 + col] = f2bf(reluf(0.9f * u6 + 0.1f * x0));
    }
  }
}

// ------------------- tk2 = relu(concat(u0,u1) @ theta) ----------------------
__global__ __launch_bounds__(128) void k_theta(const u16* __restrict__ u0b,
                                               const u16* __restrict__ u1b,
                                               const u16* __restrict__ thT,
                                               u16* __restrict__ tk2) {
  int wave = threadIdx.x >> 5;
  int lane = threadIdx.x & 31, r = lane & 15, half = lane >> 4;
  int c0 = wave * 16;
  int m0 = blockIdx.x * 16;
  int mrow = m0 + r;                     // row over (B,400,30)
  int b = mrow / 12000;
  int n = (mrow / 30) % 400;
  int t = mrow % 30;
  const u16* arow = (n < 81)
      ? u0b + ((size_t)(b * 96 + n) * 30 + t) * 64
      : u1b + ((size_t)(b * 320 + (n - 81)) * 30 + t) * 64;
  const u16* brow = thT + (c0 + r) * 64;
  v8f acc = {};
#pragma unroll
  for (int k0 = 0; k0 < 64; k0 += 32)
    acc = wmma_bf16(load_a(arow, k0, half), load_bT(brow, k0, half), acc);
  int col = c0 + r;
#pragma unroll
  for (int v = 0; v < 8; ++v)
    tk2[(size_t)(m0 + v + half * 8) * 64 + col] = f2bf(reluf(acc[v]));
}

// ------------- temporal conv 2 as K=192 GEMM + GLU gate ---------------------
// Row (b,n,t): 3 taps x 64 ch are contiguous (T-stride == 64 elems).
__global__ __launch_bounds__(128) void k_conv2(const u16* __restrict__ tk2,
                                               const u16* __restrict__ WcatT,
                                               const float* __restrict__ b1,
                                               const float* __restrict__ bc,
                                               float* __restrict__ out) {
  int wave = threadIdx.x >> 5;
  int lane = threadIdx.x & 31, r = lane & 15, half = lane >> 4;
  int c0 = wave * 16;
  int m0 = blockIdx.x * 16;
  int mrow = m0 + r;                     // row over (B,400,28)
  int b = mrow / 11200;
  int n = (mrow / 28) % 400;
  int t = mrow % 28;
  const u16* arow = tk2 + ((size_t)(b * 400 + n) * 30 + t) * 64;  // 192 contig
  const u16* w0 = WcatT + (size_t)(c0 + r) * 192;
  const u16* w1r = WcatT + (size_t)(64 + c0 + r) * 192;
  const u16* w2 = WcatT + (size_t)(128 + c0 + r) * 192;
  v8f a0 = {}, a1 = {}, a2 = {};
#pragma unroll
  for (int k0 = 0; k0 < 192; k0 += 32) {
    v16bf at = load_a(arow, k0, half);
    a0 = wmma_bf16(at, load_bT(w0, k0, half), a0);
    a1 = wmma_bf16(at, load_bT(w1r, k0, half), a1);
    a2 = wmma_bf16(at, load_bT(w2, k0, half), a2);
  }
  int col = c0 + r;
  float bt_ = b1[col], bp = bc[col], bq = bc[64 + col];
#pragma unroll
  for (int v = 0; v < 8; ++v) {
    int m = m0 + v + half * 8;
    float temp = a0[v] + bt_;
    float xp = a1[v] + bp;
    float xq = a2[v] + bq;
    out[(size_t)m * 64 + col] = (xp + temp) * sigm(xq);
  }
}

// ------------------------------- batchnorm ----------------------------------
__global__ __launch_bounds__(256) void k_bn_stats(const float* __restrict__ x,
                                                  float* __restrict__ stats) {
  int n = blockIdx.x;                    // 400 nodes
  float s = 0.f, s2 = 0.f;
  for (int i = threadIdx.x; i < 16 * 1792; i += 256) {
    int b = i / 1792, rest = i % 1792;   // 1792 = 28*64
    float v = x[(size_t)(b * 400 + n) * 1792 + rest];
    s += v; s2 += v * v;
  }
  __shared__ float sA[256], sB[256];
  sA[threadIdx.x] = s; sB[threadIdx.x] = s2;
  __syncthreads();
  for (int off = 128; off > 0; off >>= 1) {
    if (threadIdx.x < off) {
      sA[threadIdx.x] += sA[threadIdx.x + off];
      sB[threadIdx.x] += sB[threadIdx.x + off];
    }
    __syncthreads();
  }
  if (threadIdx.x == 0) {
    float mean = sA[0] / 28672.f;
    float var = sB[0] / 28672.f - mean * mean;
    stats[n] = mean;
    stats[400 + n] = var;
  }
}

__global__ void k_bn_apply(float* __restrict__ x, const float* __restrict__ stats,
                           const float* __restrict__ gamma,
                           const float* __restrict__ beta) {
  int idx = blockIdx.x * 256 + threadIdx.x;
  if (idx >= 16 * 400 * 1792) return;
  int n = (idx / 1792) % 400;
  float m = stats[n], v = stats[400 + n];
  x[idx] = (x[idx] - m) * rsqrtf(v + 1e-5f) * gamma[n] + beta[n];
}

// ------------------------------- host side ----------------------------------

static inline unsigned cdiv(unsigned a, unsigned b) { return (a + b - 1) / b; }

extern "C" void kernel_launch(void* const* d_in, const int* in_sizes, int n_in,
                              void* d_out, int out_size, void* d_ws, size_t ws_size,
                              hipStream_t stream) {
  (void)in_sizes; (void)n_in; (void)out_size; (void)ws_size;
  const float* X   = (const float*)d_in[0];
  const float* A00 = (const float*)d_in[1];
  const float* A01 = (const float*)d_in[2];
  const float* A10 = (const float*)d_in[3];
  const float* A11 = (const float*)d_in[4];
  const float* t1w = (const float*)d_in[5];
  const float* t1b = (const float*)d_in[6];
  const float* t1cw = (const float*)d_in[7];
  const float* t1cb = (const float*)d_in[8];
  const float* gth[6] = {(const float*)d_in[9],  (const float*)d_in[10],
                         (const float*)d_in[11], (const float*)d_in[12],
                         (const float*)d_in[13], (const float*)d_in[14]};
  const float* theta = (const float*)d_in[15];
  const float* t2w = (const float*)d_in[16];
  const float* t2b = (const float*)d_in[17];
  const float* t2cw = (const float*)d_in[18];
  const float* t2cb = (const float*)d_in[19];
  const float* bn_g = (const float*)d_in[20];
  const float* bn_b = (const float*)d_in[21];
  float* out = (float*)d_out;

  char* ws = (char*)d_ws;
  // big activation buffers (bf16)
  constexpr size_t OFF_T1 = 0;                       // (16,400,30,64) 24,576,000
  constexpr size_t OFF_U0 = 24576000;                // (16, 96,30,64)  5,898,240
  constexpr size_t OFF_U1 = OFF_U0 + 5898240;        // (16,320,30,64) 19,660,800
  constexpr size_t OFF_S1 = OFF_U1 + 19660800;
  constexpr size_t OFF_S2 = OFF_S1 + 19660800;
  constexpr size_t OFF_S3 = OFF_S2 + 19660800;
  constexpr size_t OFF_TK2 = OFF_S1;                 // aliases S1+S2 after layers
  constexpr size_t C0 = OFF_S3 + 19660800;           // constants region
  u16* t1bf = (u16*)(ws + OFF_T1);
  u16* u0b = (u16*)(ws + OFF_U0);
  u16* u1b = (u16*)(ws + OFF_U1);
  u16* S1 = (u16*)(ws + OFF_S1);
  u16* S2 = (u16*)(ws + OFF_S2);
  u16* S3 = (u16*)(ws + OFF_S3);
  u16* tk2 = (u16*)(ws + OFF_TK2);
  u16* A00b  = (u16*)(ws + C0 + 0);
  u16* A01b  = (u16*)(ws + C0 + 18432);
  u16* A10b  = (u16*)(ws + C0 + 79872);
  u16* A11b  = (u16*)(ws + C0 + 141312);
  u16* A01bT = (u16*)(ws + C0 + 346112);
  u16* A10bT = (u16*)(ws + C0 + 407552);
  u16* P0b   = (u16*)(ws + C0 + 468992);
  u16* P1b   = (u16*)(ws + C0 + 487424);
  u16* THT   = (u16*)(ws + C0 + 692224);   // 6 groups x 5 layers x 64x64
  u16* THETAT= (u16*)(ws + C0 + 937984);
  u16* WCATT = (u16*)(ws + C0 + 946176);
  float* STATS = (float*)(ws + C0 + 1019904);

  // zero padded activation buffers (pad node rows must stay zero)
  (void)hipMemsetAsync(u0b, 0, 5898240, stream);
  (void)hipMemsetAsync(u1b, 0, 19660800, stream);

  // ---- weight prep ----
  k_cvt_pad<<<cdiv(96 * 96, 256), 256, 0, stream>>>(A00, A00b, 81, 81, 96, 96);
  k_cvt_pad<<<cdiv(96 * 320, 256), 256, 0, stream>>>(A01, A01b, 81, 319, 96, 320);
  k_cvt_pad<<<cdiv(320 * 96, 256), 256, 0, stream>>>(A10, A10b, 319, 81, 320, 96);
  k_cvt_pad<<<cdiv(320 * 320, 256), 256, 0, stream>>>(A11, A11b, 319, 319, 320, 320);
  k_cvt_pad_tr<<<cdiv(320 * 96, 256), 256, 0, stream>>>(A01, A01bT, 81, 319, 320, 96);
  k_cvt_pad_tr<<<cdiv(96 * 320, 256), 256, 0, stream>>>(A10, A10bT, 319, 81, 96, 320);
  for (int g = 0; g < 6; ++g)
    for (int i = 0; i < 5; ++i)
      k_cvt_pad_tr<<<cdiv(64 * 64, 256), 256, 0, stream>>>(
          gth[g] + i * 4096, THT + (g * 5 + i) * 4096, 64, 64, 64, 64);
  k_cvt_pad_tr<<<cdiv(64 * 64, 256), 256, 0, stream>>>(theta, THETAT, 64, 64, 64, 64);
  k_build_wcatT<<<cdiv(192 * 192, 256), 256, 0, stream>>>(t2w, t2cw, WCATT);

  // ---- P0 = A01 @ A10 (96x96), P1 = A10 @ A01 (320x320) ----
  k_gemm_TB<<<dim3(6, 6), 32, 0, stream>>>(A01b, A10bT, P0b, 96, 320, 96);
  k_gemm_TB<<<dim3(20, 20), 32, 0, stream>>>(A10b, A01bT, P1b, 320, 96, 320);

  // ---- temporal conv 1 ----
  k_tconv1<<<cdiv(16 * 400 * 30 * 64, 256), 256, 0, stream>>>(
      X, t1w, t1b, t1cw, t1cb, t1bf, u0b, u1b);

  // ---- 5 het-conv layers ----
  for (int i = 0; i < 5; ++i) {
    // u0 update (81 nodes, padded 96)
    k_node_gemm_lds<96><<<dim3(120, 1, 16), 128, 0, stream>>>(
        A00b, P0b, u0b, S1, S2, 96);
    k_node_gemm_lds<320><<<dim3(120, 1, 16), 128, 0, stream>>>(
        A01b, nullptr, u1b, S3, nullptr, 96);
    k_het_channel<<<2880, 128, 0, stream>>>(
        S1, S2, S3, THT + (0 * 5 + i) * 4096, THT + (1 * 5 + i) * 4096,
        THT + (2 * 5 + i) * 4096, t1bf, 0, u0b, 96, 81);
    // u1 update (319 nodes, padded 320) — uses the freshly written u0
    k_node_gemm_lds<320><<<dim3(120, 1, 16), 128, 0, stream>>>(
        A11b, P1b, u1b, S1, S2, 320);
    k_node_gemm_lds<96><<<dim3(120, 1, 16), 128, 0, stream>>>(
        A10b, nullptr, u0b, S3, nullptr, 320);
    k_het_channel<<<9600, 128, 0, stream>>>(
        S1, S2, S3, THT + (3 * 5 + i) * 4096, THT + (4 * 5 + i) * 4096,
        THT + (5 * 5 + i) * 4096, t1bf, 81, u1b, 320, 319);
  }

  // ---- tk2 = relu(concat(u0,u1) @ theta) ----
  k_theta<<<12000, 128, 0, stream>>>(u0b, u1b, THETAT, tk2);

  // ---- temporal conv 2 (K=192 GEMM + gate) straight into d_out ----
  k_conv2<<<11200, 128, 0, stream>>>(tk2, WCATT, t2b, t2cb, out);

  // ---- per-node batchnorm (in place on d_out) ----
  k_bn_stats<<<400, 256, 0, stream>>>(out, STATS);
  k_bn_apply<<<cdiv(16 * 400 * 1792, 256), 256, 0, stream>>>(out, STATS, bn_g, bn_b);
}
